// ConCare_26190710571437
// MI455X (gfx1250) — compile-verified, matrix operand order checked by reading, not run on hardware
//
#include <hip/hip_runtime.h>
#include <math.h>

// ---------------------------------------------------------------------------
// Problem constants (from reference)
// ---------------------------------------------------------------------------
#define CB   128   // batch
#define CT   128   // time steps
#define CN   76    // features
#define CH   64    // hidden
#define CL   77    // tokens in transformer stage (N + 1 demo token)
#define CLP  80    // padded to 5 WMMA row tiles
#define CDFF 256

typedef __attribute__((ext_vector_type(16))) _Float16 v16h;
typedef __attribute__((ext_vector_type(8)))  float    v8f;

__device__ __forceinline__ float sigm(float x) { return 1.0f / (1.0f + __expf(-x)); }

// ---------------------------------------------------------------------------
// CDNA5 async global->LDS copy (ASYNCcnt-tracked, no VGPR data return).
// vdst = per-lane LDS byte offset; vaddr = 64-bit global address ("GV" mode).
// ---------------------------------------------------------------------------
__device__ __forceinline__ void async_load_b128(unsigned lds_off, const void* gaddr) {
  asm volatile("global_load_async_to_lds_b128 %0, %1, off"
               :: "v"(lds_off), "v"(gaddr)
               : "memory");
}
__device__ __forceinline__ void wait_asynccnt0() {
  asm volatile("s_wait_asynccnt 0" ::: "memory");
}

// ---------------------------------------------------------------------------
// WMMA fragment helpers (wave32, 16x16x32 f16 -> f32)
//
// A-matrix 16x32 (MxK), 16-bit, per CDNA5 ISA 7.12.2:
//   lanes 0-15: M=lane,    VGPR p<4 holds K = 2p,2p+1 ; p>=4 holds K = 16+2(p-4)..
//   lanes16-31: M=lane-16, same but K offset +8
// B-matrix 32x16 (KxN): mirrored layout, lane%16 selects column N.
// C/D 16x16 f32: lane%16 = N, VGPR e = row M = e + 8*(lane/16).
// ---------------------------------------------------------------------------
__device__ __forceinline__ v16h load_a16(const _Float16* __restrict__ base, int ld,
                                         int k0, int lane) {
  int m = lane & 15, hi = lane >> 4;
  const _Float16* row = base + m * ld;
  v16h r;
#pragma unroll
  for (int p = 0; p < 8; ++p) {
    int k = k0 + ((p & 3) << 1) + (hi << 3) + ((p >> 2) << 4);
    r[2 * p]     = row[k];
    r[2 * p + 1] = row[k + 1];
  }
  return r;
}

// B from an fp16 matrix stored [col][k] row-major (i.e. torch [out][in] layout)
__device__ __forceinline__ v16h load_b16(const _Float16* __restrict__ base, int ld,
                                         int k0, int lane) {
  int c = lane & 15, hi = lane >> 4;
  const _Float16* row = base + c * ld;
  v16h r;
#pragma unroll
  for (int p = 0; p < 8; ++p) {
    int k = k0 + ((p & 3) << 1) + (hi << 3) + ((p >> 2) << 4);
    r[2 * p]     = row[k];
    r[2 * p + 1] = row[k + 1];
  }
  return r;
}

// B from an fp32 weight matrix in torch layout W[out][in] (convert on load)
__device__ __forceinline__ v16h load_b32f(const float* __restrict__ base, int ld,
                                          int k0, int lane) {
  int c = lane & 15, hi = lane >> 4;
  const float* row = base + c * ld;
  v16h r;
#pragma unroll
  for (int p = 0; p < 8; ++p) {
    int k = k0 + ((p & 3) << 1) + (hi << 3) + ((p >> 2) << 4);
    r[2 * p]     = (_Float16)row[k];
    r[2 * p + 1] = (_Float16)row[k + 1];
  }
  return r;
}

// Generic wave-cooperative GEMM: out[f16, Mpad x Ncols] = A[f16, Mpad x K] @ W^T + bias
// W is fp32 torch-layout [Ncols][ldw>=K]. Each wave owns a strided set of 16x16 tiles.
// C starts as the WMMA inline-constant 0; bias / accumulate value is added in the
// store epilogue (avoids per-tile splat movs on the WMMA critical path).
__device__ void wave_gemm(const _Float16* __restrict__ A, int lda,
                          const float* __restrict__ W, int ldw,
                          const float* __restrict__ bias,
                          _Float16* __restrict__ out, int ldo,
                          int Mtiles, int Ntiles, int K,
                          bool relu, bool accum,
                          int wave, int nwaves, int lane) {
  int col_l = lane & 15;
  int row_l = (lane >> 4) << 3;
  for (int t = wave; t < Mtiles * Ntiles; t += nwaves) {
    int mt = t / Ntiles, nt = t - mt * Ntiles;
    int col = nt * 16 + col_l;
    int rb  = mt * 16 + row_l;
    v8f c = {};  // inline-0 C operand on first WMMA
    for (int k0 = 0; k0 < K; k0 += 32) {
      v16h a = load_a16(A + mt * 16 * lda, lda, k0, lane);
      v16h b = load_b32f(W + nt * 16 * ldw, ldw, k0, lane);
      c = __builtin_amdgcn_wmma_f32_16x16x32_f16(false, a, false, b, (short)0, c,
                                                 false, false);
    }
    float bv = bias ? bias[col] : 0.0f;
#pragma unroll
    for (int e = 0; e < 8; ++e) {
      float base = accum ? (float)out[(rb + e) * ldo + col] : bv;
      float v = c[e] + base;
      if (relu) v = fmaxf(v, 0.0f);
      out[(rb + e) * ldo + col] = (_Float16)v;
    }
  }
}

// ---------------------------------------------------------------------------
// Kernel 1: per-feature GRU over T steps. One block per feature n.
// 8 waves x 16 batch rows; hh-GEMM [128x64]@[64x192] via WMMA each step.
// C-fragments start from the inline-0 operand; bhh is folded into the gate
// epilogue (r/z: pre-summed with bih; n: added inside the r* product).
// Writes Hs as fp16 [B][N][T][H].
// ---------------------------------------------------------------------------
__global__ __launch_bounds__(256) void gru_kernel(
    const float* __restrict__ X,    // [B,T,N]
    const float* __restrict__ Wih,  // [N,3H]
    const float* __restrict__ Whh,  // [N,3H,H]
    const float* __restrict__ bih,  // [N,3H]
    const float* __restrict__ bhh,  // [N,3H]
    _Float16* __restrict__ Hs)      // [B,N,T,H] fp16
{
  __shared__ __align__(16) _Float16 sW[3 * CH * CH];  // Whh[n] as [col(192)][k(64)]
  __shared__ float sWih[3 * CH];
  __shared__ float sBrz[2 * CH];   // bih+bhh for r,z columns
  __shared__ float sBnih[CH];      // bih for n columns
  __shared__ float sBnhh[CH];      // bhh for n columns
  __shared__ float sX[CB];
  __shared__ __align__(16) _Float16 sH[8][16 * CH];   // per-wave h (fp16, [m][k])

  const int n    = blockIdx.x;
  const int tid  = threadIdx.x;
  const int lane = tid & 31;
  const int wave = tid >> 5;

  for (int i = tid; i < 3 * CH * CH; i += 256) sW[i] = (_Float16)Whh[(size_t)n * 3 * CH * CH + i];
  for (int i = tid; i < 3 * CH; i += 256) sWih[i] = Wih[n * 3 * CH + i];
  for (int i = tid; i < 2 * CH; i += 256) sBrz[i] = bih[n * 3 * CH + i] + bhh[n * 3 * CH + i];
  for (int i = tid; i < CH; i += 256) {
    sBnih[i] = bih[n * 3 * CH + 2 * CH + i];
    sBnhh[i] = bhh[n * 3 * CH + 2 * CH + i];
  }
  for (int i = tid; i < 8 * 16 * CH; i += 256) ((_Float16*)sH)[i] = (_Float16)0.0f;
  __syncthreads();

  const int colL = lane & 15;
  const int hi   = lane >> 4;
  float hst[4][8];
#pragma unroll
  for (int a = 0; a < 4; ++a)
#pragma unroll
    for (int e = 0; e < 8; ++e) hst[a][e] = 0.0f;

  for (int t = 0; t < CT; ++t) {
    if (tid < CB) sX[tid] = X[(size_t)tid * CT * CN + (size_t)t * CN + n];
    __syncthreads();

    // A fragments of current h (two K=32 halves)
    v16h a0 = load_a16(sH[wave], CH, 0, lane);
    v16h a1 = load_a16(sH[wave], CH, 32, lane);

    v8f acc[12];
#pragma unroll
    for (int tc = 0; tc < 12; ++tc) {
      v8f c = {};  // inline-0 C
      v16h b0 = load_b16(sW + tc * 16 * CH, CH, 0, lane);
      c = __builtin_amdgcn_wmma_f32_16x16x32_f16(false, a0, false, b0, (short)0, c, false, false);
      v16h b1 = load_b16(sW + tc * 16 * CH, CH, 32, lane);
      c = __builtin_amdgcn_wmma_f32_16x16x32_f16(false, a1, false, b1, (short)0, c, false, false);
      acc[tc] = c;
    }

    // gates (PyTorch order r,z,n); C-layout: lane holds col j, rows m = e + 8*hi
#pragma unroll
    for (int t4 = 0; t4 < 4; ++t4) {
      int j = t4 * 16 + colL;
      float wr = sWih[j], wz = sWih[CH + j], wn = sWih[2 * CH + j];
      float br = sBrz[j], bz = sBrz[CH + j];
      float bnih = sBnih[j], bnhh = sBnhh[j];
#pragma unroll
      for (int e = 0; e < 8; ++e) {
        int m = e + (hi << 3);
        float x  = sX[wave * 16 + m];
        float rg = sigm(x * wr + br + acc[t4][e]);
        float zg = sigm(x * wz + bz + acc[t4 + 4][e]);
        float ng = tanhf(x * wn + bnih + rg * (acc[t4 + 8][e] + bnhh));
        float hn = (1.0f - zg) * ng + zg * hst[t4][e];
        hst[t4][e] = hn;
        sH[wave][m * CH + j] = (_Float16)hn;
      }
    }
    __syncthreads();  // make sH visible across lanes/waves before re-reading

    // coalesced Hs store: lane handles row m=lane&15, half=(lane>>4): 32 halves = 64B
    {
      int m   = lane & 15;
      int off = (lane >> 4) * 32;
      int bg  = wave * 16 + m;
      const uint4* src = (const uint4*)(&sH[wave][m * CH + off]);
      uint4* dst = (uint4*)(Hs + ((((size_t)bg * CN + n) * CT + t) * CH + off));
#pragma unroll
      for (int q = 0; q < 4; ++q) dst[q] = src[q];
    }
    __syncthreads();
  }
}

// ---------------------------------------------------------------------------
// Kernel 2: time attention per (b,n). Folds Wx into the query so the full K
// tensor is never materialized (saves ~1e10 FLOPs and 320MB of traffic).
// The 16KB Hs tile is staged into LDS ONCE via CDNA5 async global->LDS loads
// (ASYNCcnt) and served from LDS for all three passes (~3x HBM traffic cut on
// the dominant 152MB stream).
// ---------------------------------------------------------------------------
__global__ __launch_bounds__(128) void attn_time_kernel(
    const _Float16* __restrict__ Hs, const float* __restrict__ Wt,
    const float* __restrict__ bt, const float* __restrict__ Wx,
    const float* __restrict__ bx, const float* __restrict__ rate,
    float* __restrict__ F_in)  // [B,77,64]
{
  const int b = blockIdx.x / CN;
  const int n = blockIdx.x % CN;
  const _Float16* Hbn = Hs + (((size_t)b * CN + n) * CT) * CH;

  __shared__ __align__(16) _Float16 sT[CT * CH];  // 16KB tile of Hs[b,n]
  __shared__ float hl[CH], q[CH], qt[CH], ev[CT], red[CT];
  __shared__ float c0s;
  const int tid = threadIdx.x;

  // async-stage the tile: 1024 x 16B chunks, 8 per thread
  {
    unsigned lds_base = (unsigned)(uintptr_t)(&sT[0]);
    const char* g = (const char*)Hbn;
#pragma unroll
    for (int i = 0; i < 8; ++i) {
      int off = (tid + i * 128) * 16;
      async_load_b128(lds_base + off, g + off);
    }
    wait_asynccnt0();
  }
  __syncthreads();

  if (tid < CH) hl[tid] = (float)sT[(CT - 1) * CH + tid];
  __syncthreads();

  if (tid < CH) {
    const float* wrow = Wt + ((size_t)n * CH + tid) * CH;
    float a = bt[n * CH + tid];
    for (int h = 0; h < CH; ++h) a += hl[h] * wrow[h];
    q[tid] = a;
  }
  __syncthreads();

  if (tid < CH) {
    float a = 0.0f;
    for (int k = 0; k < CH; ++k) a += q[k] * Wx[((size_t)n * CH + k) * CH + tid];
    qt[tid] = a;
  }
  if (tid == 0) {
    float a = 0.0f;
    for (int k = 0; k < CH; ++k) a += q[k] * bx[n * CH + k];
    c0s = a;
  }
  __syncthreads();

  const float srate = sigm(rate[n]);
  {
    int t = tid;  // 128 threads == T
    float d = c0s;
    for (int h = 0; h < CH; ++h) d += qt[h] * (float)sT[t * CH + h];
    float ds  = sigm(d);
    float den = srate * __logf(2.72f + (1.0f - ds)) * (float)(t + 1);
    den = fmaxf(den, 1e-6f);
    ev[t] = fmaxf(ds / den, 0.0f);
  }
  __syncthreads();

  red[tid] = ev[tid];
  __syncthreads();
  for (int s = 64; s >= 1; s >>= 1) {
    if (tid < s) red[tid] = fmaxf(red[tid], red[tid + s]);
    __syncthreads();
  }
  float mx = red[0];
  __syncthreads();
  float ex = __expf(ev[tid] - mx);
  ev[tid] = ex;
  red[tid] = ex;
  __syncthreads();
  for (int s = 64; s >= 1; s >>= 1) {
    if (tid < s) red[tid] += red[tid + s];
    __syncthreads();
  }
  float inv = 1.0f / red[0];
  __syncthreads();

  if (tid < CH) {
    float a = 0.0f;
    for (int t = 0; t < CT; ++t) a += ev[t] * (float)sT[t * CH + tid];
    F_in[((size_t)b * CL + n) * CH + tid] = a * inv;
  }
}

// ---------------------------------------------------------------------------
// Kernel 3: demographic token
// ---------------------------------------------------------------------------
__global__ void demo_kernel(const float* __restrict__ D, const float* __restrict__ W,
                            const float* __restrict__ bias, float* __restrict__ F_in) {
  int b = blockIdx.x, h = threadIdx.x;  // 64 threads
  float a = bias[h];
  for (int d = 0; d < 12; ++d) a += D[b * 12 + d] * W[h * 12 + d];
  F_in[((size_t)b * CL + CN) * CH + h] = tanhf(a);
}

// ---------------------------------------------------------------------------
// Kernel 4: fused transformer stage per batch element.
// ---------------------------------------------------------------------------
struct K4Args {
  const float *F_in;
  const float *Wq, *bq, *Wk, *bk, *Wv, *bv, *Wo, *bo;
  const float *g1, *b1;
  const float *fW1, *fb1, *fW2, *fb2;
  const float *g2, *b2;
  const float *faWq, *fabq, *faWk, *fabk, *faWv, *fabv, *faWout, *fabout;
  const float *o0W, *o0b, *o1W, *o1b;
  float *S1, *S2, *yout;
};

__device__ void mha_core(_Float16* sA, _Float16* sQ, _Float16* sK, _Float16* sV,
                         _Float16* sc, const K4Args& P, int tid, int wave, int lane) {
  wave_gemm(sA, CH, P.Wq, CH, P.bq, sQ, CH, 5, 4, CH, false, false, wave, 8, lane);
  wave_gemm(sA, CH, P.Wk, CH, P.bk, sK, CH, 5, 4, CH, false, false, wave, 8, lane);
  wave_gemm(sA, CH, P.Wv, CH, P.bv, sV, CH, 5, 4, CH, false, false, wave, 8, lane);
  __syncthreads();
  for (int hd = 0; hd < 4; ++hd) {
    const int ho = hd * 16;
    for (int idx = tid; idx < CL * CL; idx += 256) {
      int l1 = idx / CL, l2 = idx - l1 * CL;
      float a = 0.0f;
#pragma unroll
      for (int j = 0; j < 16; ++j)
        a += (float)sQ[l1 * CH + ho + j] * (float)sK[l2 * CH + ho + j];
      sc[l1 * CLP + l2] = (_Float16)(a * 0.25f);  // 1/sqrt(16)
    }
    __syncthreads();
    for (int l1 = tid; l1 < CL; l1 += 256) {
      float mx = -1e30f;
      for (int l2 = 0; l2 < CL; ++l2) mx = fmaxf(mx, (float)sc[l1 * CLP + l2]);
      float s = 0.0f;
      for (int l2 = 0; l2 < CL; ++l2) {
        float e = __expf((float)sc[l1 * CLP + l2] - mx);
        s += e;
        sc[l1 * CLP + l2] = (_Float16)e;
      }
      float inv = 1.0f / s;
      for (int l2 = 0; l2 < CL; ++l2)
        sc[l1 * CLP + l2] = (_Float16)((float)sc[l1 * CLP + l2] * inv);
    }
    __syncthreads();
    for (int idx = tid; idx < CL * 16; idx += 256) {
      int l1 = idx >> 4, j = idx & 15;
      float a = 0.0f;
      for (int l2 = 0; l2 < CL; ++l2)
        a += (float)sc[l1 * CLP + l2] * (float)sV[l2 * CH + ho + j];
      sA[l1 * CH + ho + j] = (_Float16)a;  // merged heads overwrite input (dead)
    }
    __syncthreads();
  }
  for (int idx = tid; idx < 3 * CH; idx += 256)
    sA[(CL + idx / CH) * CH + (idx & 63)] = (_Float16)0.0f;  // zero pad rows
  __syncthreads();
}

__global__ __launch_bounds__(256) void fuse_kernel(K4Args P) {
  // 61440B overlayed arena + small scratch
  __shared__ __align__(16) char smem[10240 * 4 + 20480];
  _Float16* sA = (_Float16*)smem;   // 80x64 f16: gemm input / merged heads
  _Float16* sQ = sA + CLP * CH;     // 80x64 f16
  _Float16* sK = sQ + CLP * CH;     // 80x64 f16
  _Float16* sV = sK + CLP * CH;     // 80x64 f16
  _Float16* sR = sV + CLP * CH;     // 80x128 f16: scores(ld 80) / ffn hidden(ld 128)
  float*    Yf = (float*)sK;        // 80x64 f32 overlays sK+sV (phase-disjoint)
  __shared__ float fsc[CLP], svec[CH], rbuf[256];

  const int b    = blockIdx.x;
  const int tid  = threadIdx.x;
  const int lane = tid & 31;
  const int wave = tid >> 5;
  const float* Fin = P.F_in + (size_t)b * CL * CH;

  // ---- pass 0: MHA on raw F_in, only for decov statistics ----
  for (int idx = tid; idx < CLP * CH; idx += 256) {
    int l = idx >> 6;
    sA[idx] = (_Float16)(l < CL ? Fin[idx] : 0.0f);
  }
  __syncthreads();
  mha_core(sA, sQ, sK, sV, sR, P, tid, wave, lane);
  if (tid < CH) {
    float a = 0.0f;
    for (int l = 0; l < CL; ++l) a += (float)sA[l * CH + tid];
    atomicAdd(P.S1 + tid, a);
  }
  for (int idx = tid; idx < CH * CH; idx += 256) {
    int i = idx >> 6, j = idx & 63;
    float a = 0.0f;
    for (int l = 0; l < CL; ++l) a += (float)sA[l * CH + i] * (float)sA[l * CH + j];
    atomicAdd(P.S2 + idx, a);
  }
  __syncthreads();

  // ---- pass 1: MHA on layernorm(F_in) ----
  for (int l = tid; l < CLP; l += 256) {
    if (l < CL) {
      float mu = 0.0f;
      for (int h = 0; h < CH; ++h) mu += Fin[l * CH + h];
      mu *= (1.0f / CH);
      float var = 0.0f;
      for (int h = 0; h < CH; ++h) { float d = Fin[l * CH + h] - mu; var += d * d; }
      float inv = rsqrtf(var * (1.0f / CH) + 1e-7f);
      for (int h = 0; h < CH; ++h)
        sA[l * CH + h] = (_Float16)((Fin[l * CH + h] - mu) * inv * P.g1[h] + P.b1[h]);
    } else {
      for (int h = 0; h < CH; ++h) sA[l * CH + h] = (_Float16)0.0f;
    }
  }
  __syncthreads();
  mha_core(sA, sQ, sK, sV, sR, P, tid, wave, lane);
  wave_gemm(sA, CH, P.Wo, CH, P.bo, sQ, CH, 5, 4, CH, false, false, wave, 8, lane);
  __syncthreads();

  // ---- residual Y = F_in + attn_out (fp32, overlays sK/sV) ----
  for (int idx = tid; idx < CLP * CH; idx += 256) {
    int l = idx >> 6;
    Yf[idx] = (l < CL) ? (Fin[idx] + (float)sQ[idx]) : 0.0f;
  }
  __syncthreads();

  // ---- ln2(Y) -> sA ----
  for (int l = tid; l < CLP; l += 256) {
    if (l < CL) {
      float mu = 0.0f;
      for (int h = 0; h < CH; ++h) mu += Yf[l * CH + h];
      mu *= (1.0f / CH);
      float var = 0.0f;
      for (int h = 0; h < CH; ++h) { float d = Yf[l * CH + h] - mu; var += d * d; }
      float inv = rsqrtf(var * (1.0f / CH) + 1e-7f);
      for (int h = 0; h < CH; ++h)
        sA[l * CH + h] = (_Float16)((Yf[l * CH + h] - mu) * inv * P.g2[h] + P.b2[h]);
    } else {
      for (int h = 0; h < CH; ++h) sA[l * CH + h] = (_Float16)0.0f;
    }
  }
  __syncthreads();

  // ---- FFN in two 128-column halves; second W2 gemm accumulates ----
  for (int half = 0; half < 2; ++half) {
    wave_gemm(sA, CH, P.fW1 + half * 128 * CH, CH, P.fb1 + half * 128,
              sR, 128, 5, 8, CH, true, false, wave, 8, lane);
    __syncthreads();
    wave_gemm(sR, 128, P.fW2 + half * 128, CDFF, P.fb2,
              sQ, CH, 5, 4, 128, false, half == 1, wave, 8, lane);
    __syncthreads();
  }

  // ---- Z = Y + ffn_out -> sA (f16) ----
  for (int idx = tid; idx < CLP * CH; idx += 256) {
    int l = idx >> 6;
    sA[idx] = (_Float16)(l < CL ? (Yf[idx] + (float)sQ[idx]) : 0.0f);
  }
  __syncthreads();

  // ---- FinalAttentionQKV ----
  wave_gemm(sA, CH, P.faWq, CH, P.fabq, sQ, CH, 5, 4, CH, false, false, wave, 8, lane);
  wave_gemm(sA, CH, P.faWk, CH, P.fabk, sK, CH, 5, 4, CH, false, false, wave, 8, lane);
  wave_gemm(sA, CH, P.faWv, CH, P.fabv, sV, CH, 5, 4, CH, false, false, wave, 8, lane);
  __syncthreads();

  for (int l = tid; l < CL; l += 256) {
    float a = P.fabout[0];
    for (int h = 0; h < CH; ++h)
      a += (float)sQ[l * CH + h] * (float)sK[l * CH + h] * P.faWout[h];
    fsc[l] = a;
  }
  __syncthreads();
  if (tid == 0) {
    float mx = -1e30f;
    for (int l = 0; l < CL; ++l) mx = fmaxf(mx, fsc[l]);
    rbuf[0] = mx;
  }
  __syncthreads();
  float mx = rbuf[0];
  for (int l = tid; l < CL; l += 256) fsc[l] = __expf(fsc[l] - mx);
  __syncthreads();
  if (tid == 0) {
    float s = 0.0f;
    for (int l = 0; l < CL; ++l) s += fsc[l];
    rbuf[1] = 1.0f / s;
  }
  __syncthreads();
  if (tid < CH) {
    float a = 0.0f;
    for (int l = 0; l < CL; ++l) a += fsc[l] * (float)sV[l * CH + tid];
    svec[tid] = a * rbuf[1];
  }
  __syncthreads();
  if (tid < CH) {
    float a = P.o0b[tid];
    for (int k = 0; k < CH; ++k) a += svec[k] * P.o0W[tid * CH + k];
    rbuf[tid] = fmaxf(a, 0.0f) * P.o1W[tid];
  }
  __syncthreads();
  if (tid == 0) {
    float a = P.o1b[0];
    for (int h = 0; h < CH; ++h) a += rbuf[h];
    P.yout[b] = sigm(a);
  }
}

// ---------------------------------------------------------------------------
// Kernel 5: finalize decov from accumulated S1/S2
// ---------------------------------------------------------------------------
__global__ void decov_kernel(const float* __restrict__ S1, const float* __restrict__ S2,
                             float* __restrict__ out) {
  __shared__ float r1[256], r2[256];
  const int tid = threadIdx.x;
  const float M = (float)(CB * CL);
  float s = 0.0f, d = 0.0f;
  for (int idx = tid; idx < CH * CH; idx += 256) {
    int i = idx >> 6, j = idx & 63;
    float C = S2[idx] / M - (S1[i] / M) * (S1[j] / M);
    s += C * C;
    if (i == j) d += C * C;
  }
  r1[tid] = s;
  r2[tid] = d;
  __syncthreads();
  for (int st = 128; st >= 1; st >>= 1) {
    if (tid < st) { r1[tid] += r1[tid + st]; r2[tid] += r2[tid + st]; }
    __syncthreads();
  }
  if (tid == 0) out[CB] = 0.5f * (r1[0] - r2[0]);
}

__global__ void zero_kernel(float* __restrict__ p, int n) {
  int i = blockIdx.x * blockDim.x + threadIdx.x;
  if (i < n) p[i] = 0.0f;
}

// ---------------------------------------------------------------------------
// Host launch
// ---------------------------------------------------------------------------
extern "C" void kernel_launch(void* const* d_in, const int* in_sizes, int n_in,
                              void* d_out, int out_size, void* d_ws, size_t ws_size,
                              hipStream_t stream) {
  (void)in_sizes; (void)n_in; (void)out_size; (void)ws_size;

  const float* X     = (const float*)d_in[0];
  const float* D     = (const float*)d_in[1];
  const float* gWih  = (const float*)d_in[2];
  const float* gWhh  = (const float*)d_in[3];
  const float* gbih  = (const float*)d_in[4];
  const float* gbhh  = (const float*)d_in[5];
  const float* aWt   = (const float*)d_in[6];
  const float* abt   = (const float*)d_in[7];
  const float* aWx   = (const float*)d_in[8];
  const float* abx   = (const float*)d_in[9];
  const float* arate = (const float*)d_in[10];
  const float* dW    = (const float*)d_in[11];
  const float* db    = (const float*)d_in[12];

  // workspace layout
  _Float16* Hs = (_Float16*)d_ws;                       // fp16 [B,N,T,H] (~152MB)
  size_t hsElems = (size_t)CB * CN * CT * CH;
  float* F_in = (float*)((char*)d_ws + hsElems * sizeof(_Float16));  // [B,77,64]
  float* S1 = F_in + (size_t)CB * CL * CH;              // [64]
  float* S2 = S1 + CH;                                  // [64*64]

  zero_kernel<<<(CH + CH * CH + 255) / 256, 256, 0, stream>>>(S1, CH + CH * CH);
  gru_kernel<<<CN, 256, 0, stream>>>(X, gWih, gWhh, gbih, gbhh, Hs);
  attn_time_kernel<<<CB * CN, 128, 0, stream>>>(Hs, aWt, abt, aWx, abx, arate, F_in);
  demo_kernel<<<CB, 64, 0, stream>>>(D, dW, db, F_in);

  K4Args P;
  P.F_in = F_in;
  P.Wq = (const float*)d_in[13];  P.bq = (const float*)d_in[14];
  P.Wk = (const float*)d_in[15];  P.bk = (const float*)d_in[16];
  P.Wv = (const float*)d_in[17];  P.bv = (const float*)d_in[18];
  P.Wo = (const float*)d_in[19];  P.bo = (const float*)d_in[20];
  P.g1 = (const float*)d_in[21];  P.b1 = (const float*)d_in[22];
  P.fW1 = (const float*)d_in[23]; P.fb1 = (const float*)d_in[24];
  P.fW2 = (const float*)d_in[25]; P.fb2 = (const float*)d_in[26];
  P.g2 = (const float*)d_in[27];  P.b2 = (const float*)d_in[28];
  P.faWq = (const float*)d_in[29];   P.fabq = (const float*)d_in[30];
  P.faWk = (const float*)d_in[31];   P.fabk = (const float*)d_in[32];
  P.faWv = (const float*)d_in[33];   P.fabv = (const float*)d_in[34];
  P.faWout = (const float*)d_in[35]; P.fabout = (const float*)d_in[36];
  P.o0W = (const float*)d_in[37];    P.o0b = (const float*)d_in[38];
  P.o1W = (const float*)d_in[39];    P.o1b = (const float*)d_in[40];
  P.S1 = S1; P.S2 = S2; P.yout = (float*)d_out;

  fuse_kernel<<<CB, 256, 0, stream>>>(P);
  decov_kernel<<<1, 256, 0, stream>>>(S1, S2, (float*)d_out);
}
